// MQAttention_62689342652872
// MI455X (gfx1250) — matrix operand order repaired and checked by reading, not compiled, not run on hardware
//
#include <hip/hip_runtime.h>
#include <hip/hip_bf16.h>

// MQ-attention (query-axis softmax) on gfx1250, bf16 WMMA, double-buffered LDS.
//
//   K1: s[b,q,k]  = (u[b,q,:] . m[b,k,:]) / 32          fp32 -> ws (134 MB, L2-resident)
//   K2: p[b,q,k]  = softmax over q of s[b,:,k]          bf16 -> ws (67 MB)
//   K3: t[b,q,d]  = bf16( p[b,q,:] . c[b,:,d] + u )     bf16 -> ws (32 MB)
//   K4: out[b,q,e]= t[b,q,:] . W[e,:] + bias[e]         fp32 -> d_out
//
// Workspace requirement: 134,217,728 + 67,108,864 + 33,554,432 = 234,881,024 B.

typedef __bf16 bf16;
typedef __attribute__((ext_vector_type(4)))  __bf16 v4bf;
typedef __attribute__((ext_vector_type(8)))  __bf16 v8bf;
typedef __attribute__((ext_vector_type(16))) __bf16 v16bf;
typedef __attribute__((ext_vector_type(8)))  float  v8f;

static constexpr int B_  = 8;
static constexpr int SQ_ = 2048;
static constexpr int SK_ = 2048;
static constexpr int D_  = 1024;
static constexpr int LDT = 72;   // LDS row stride in bf16: 64 + 8 pad (16B-aligned rows)

union F16x16 { v16bf v; v8bf h[2]; };

// ---- global->register prefetch + register->LDS stores (256 threads) -------

struct PrefF { float4 v[8]; };   // one 128x64 fp32 tile (or 64x128 for transpose)
struct PrefB { v8bf  v[4]; };    // one 128x64 bf16 tile

// 128 rows x 64 cols fp32, row stride gs.
__device__ inline PrefF gload_f32(const float* __restrict__ g, int gs, int t) {
  PrefF P;
#pragma unroll
  for (int i = 0; i < 8; ++i) {
    int idx = t + i * 256;              // 2048 float4 slots = 128 * 16
    int r = idx >> 4, c = (idx & 15) << 2;
    P.v[i] = *(const float4*)(g + (size_t)r * gs + c);
  }
  return P;
}
__device__ inline void lstore_f32(bf16* lds, const PrefF& P, int t) {
#pragma unroll
  for (int i = 0; i < 8; ++i) {
    int idx = t + i * 256;
    int r = idx >> 4, c = (idx & 15) << 2;
    v4bf o; o[0] = (bf16)P.v[i].x; o[1] = (bf16)P.v[i].y;
    o[2] = (bf16)P.v[i].z; o[3] = (bf16)P.v[i].w;
    *(v4bf*)(lds + r * LDT + c) = o;
  }
}

// 128 rows x 64 cols bf16, row stride gs.
__device__ inline PrefB gload_bf16(const bf16* __restrict__ g, int gs, int t) {
  PrefB P;
#pragma unroll
  for (int i = 0; i < 4; ++i) {
    int idx = t + i * 256;              // 1024 8-elem chunks = 128 * 8
    int r = idx >> 3, c = (idx & 7) << 3;
    P.v[i] = *(const v8bf*)(g + (size_t)r * gs + c);
  }
  return P;
}
__device__ inline void lstore_bf16(bf16* lds, const PrefB& P, int t) {
#pragma unroll
  for (int i = 0; i < 4; ++i) {
    int idx = t + i * 256;
    int r = idx >> 3, c = (idx & 7) << 3;
    *(v8bf*)(lds + r * LDT + c) = P.v[i];
  }
}

// 64(k) x 128(d) fp32 tile, row stride gs; stored transposed: lds[d][k].
__device__ inline PrefF gload_T_f32(const float* __restrict__ g, int gs, int t) {
  PrefF P;
#pragma unroll
  for (int i = 0; i < 8; ++i) {
    int idx = t + i * 256;              // 2048 float4 slots = 64 * 32
    int r = idx >> 5, c = (idx & 31) << 2;
    P.v[i] = *(const float4*)(g + (size_t)r * gs + c);
  }
  return P;
}
__device__ inline void lstore_T(bf16* lds, const PrefF& P, int t) {
#pragma unroll
  for (int i = 0; i < 8; ++i) {
    int idx = t + i * 256;
    int r = idx >> 5, c = (idx & 31) << 2;
    float vv[4] = { P.v[i].x, P.v[i].y, P.v[i].z, P.v[i].w };
#pragma unroll
    for (int j = 0; j < 4; ++j) lds[(c + j) * LDT + r] = (bf16)vv[j];
  }
}

// ---- WMMA fragment loads (ISA 7.12.2 16-bit layouts) ----------------------

// A 16x32: lanes 0-15 hold row M=l, K {0..7,16..23}; lanes 16-31 K {8..15,24..31}.
__device__ inline v16bf load_frag_a(const bf16* lds, int row0, int kk, int lane) {
  int l = lane & 15, h = lane >> 4;
  const bf16* p = lds + (row0 + l) * LDT + kk + h * 8;
  F16x16 f;
  f.h[0] = *(const v8bf*)p;
  f.h[1] = *(const v8bf*)(p + 16);
  return f.v;
}

// B 32x16: lanes 0-15 hold col N=l, K {0..7,16..23}; lanes 16-31 K {8..15,24..31}.
__device__ inline v16bf load_frag_b(const bf16* lds, int col0, int kk, int lane) {
  int l = lane & 15, h = lane >> 4;
  const bf16* p = lds + (col0 + l) * LDT + kk + h * 8;
  F16x16 f;
  f.h[0] = *(const v8bf*)p;
  f.h[1] = *(const v8bf*)(p + 16);
  return f.v;
}

// One 64-deep K chunk of a 64x32 wave tile: 2 k-substeps x (4x2) wmma.
__device__ inline void wave_mma(const bf16* lA, const bf16* lB, int wm, int wn,
                                int lane, v8f acc[4][2]) {
#pragma unroll
  for (int kk = 0; kk < 64; kk += 32) {
    v16bf a[4], b[2];
#pragma unroll
    for (int mi = 0; mi < 4; ++mi) a[mi] = load_frag_a(lA, wm + mi * 16, kk, lane);
#pragma unroll
    for (int ni = 0; ni < 2; ++ni) b[ni] = load_frag_b(lB, wn + ni * 16, kk, lane);
#pragma unroll
    for (int mi = 0; mi < 4; ++mi)
#pragma unroll
      for (int ni = 0; ni < 2; ++ni)
        acc[mi][ni] = __builtin_amdgcn_wmma_f32_16x16x32_bf16(
            false, a[mi], false, b[ni], (short)0, acc[mi][ni], false, false);
  }
}

// ---- K1: scores = u . m^T / 32 -------------------------------------------

__global__ __launch_bounds__(256) void mqa_k1_scores(
    const float* __restrict__ u, const float* __restrict__ m, float* __restrict__ s) {
  const int b  = blockIdx.z;
  const int q0 = blockIdx.x * 128, k0 = blockIdx.y * 128;
  const float* ub = u + (size_t)b * SQ_ * D_ + (size_t)q0 * D_;
  const float* mb = m + (size_t)b * SK_ * D_ + (size_t)k0 * D_;
  float* sb = s + (size_t)b * SQ_ * SK_;

  __shared__ bf16 lA[2][128 * LDT];
  __shared__ bf16 lB[2][128 * LDT];
  const int t = threadIdx.x, lane = t & 31, w = t >> 5;
  const int wm = (w & 1) * 64, wn = (w >> 1) * 32;

  PrefF pa = gload_f32(ub, D_, t);
  PrefF pb = gload_f32(mb, D_, t);
  lstore_f32(lA[0], pa, t);
  lstore_f32(lB[0], pb, t);

  v8f acc[4][2] = {};
  constexpr int NS = D_ / 64;
  for (int it = 0; it < NS; ++it) {
    __syncthreads();
    const int cur = it & 1;
    if (it + 1 < NS) {
      pa = gload_f32(ub + (it + 1) * 64, D_, t);
      pb = gload_f32(mb + (it + 1) * 64, D_, t);
    }
    wave_mma(lA[cur], lB[cur], wm, wn, lane, acc);
    if (it + 1 < NS) {
      lstore_f32(lA[cur ^ 1], pa, t);
      lstore_f32(lB[cur ^ 1], pb, t);
    }
  }

  const int l = lane & 15, h = lane >> 4;
#pragma unroll
  for (int mi = 0; mi < 4; ++mi)
#pragma unroll
    for (int ni = 0; ni < 2; ++ni)
#pragma unroll
      for (int r = 0; r < 8; ++r)
        sb[(size_t)(q0 + wm + mi * 16 + h * 8 + r) * SK_ + (k0 + wn + ni * 16 + l)] =
            acc[mi][ni][r] * 0.03125f;
}

// ---- K2: column softmax (over q) -> bf16 p --------------------------------
// 8 threads cooperate per column (online max/sum), flat LDS combine of the 8 partials, 1 write pass.

__global__ __launch_bounds__(256) void mqa_k2_softmax(
    const float* __restrict__ s, bf16* __restrict__ p) {
  const int b    = blockIdx.y;
  const int col  = blockIdx.x * 32 + (threadIdx.x & 31);
  const int part = threadIdx.x >> 5;                  // 0..7
  const float* sb = s + (size_t)b * SQ_ * SK_ + col;
  bf16* pb = p + (size_t)b * SQ_ * SK_ + col;

  float mx = -3.402823466e38f, sum = 0.f;
#pragma unroll 4
  for (int q = part; q < SQ_; q += 8) {
    const float v  = sb[(size_t)q * SK_];
    const float nm = fmaxf(mx, v);
    sum = sum * __expf(mx - nm) + __expf(v - nm);
    mx  = nm;
  }

  __shared__ float smx[8][32];
  __shared__ float ssm[8][32];
  smx[part][threadIdx.x & 31] = mx;
  ssm[part][threadIdx.x & 31] = sum;
  __syncthreads();

  float M = -3.402823466e38f;
#pragma unroll
  for (int i = 0; i < 8; ++i) M = fmaxf(M, smx[i][threadIdx.x & 31]);
  float S = 0.f;
#pragma unroll
  for (int i = 0; i < 8; ++i)
    S += ssm[i][threadIdx.x & 31] * __expf(smx[i][threadIdx.x & 31] - M);
  const float inv = 1.f / S;

#pragma unroll 4
  for (int q = part; q < SQ_; q += 8)
    pb[(size_t)q * SK_] = (bf16)(__expf(sb[(size_t)q * SK_] - M) * inv);
}

// ---- K3: t = bf16( p . c + u ) --------------------------------------------

__global__ __launch_bounds__(256) void mqa_k3_ov(
    const bf16* __restrict__ p, const float* __restrict__ c,
    const float* __restrict__ u, bf16* __restrict__ tt) {
  const int b  = blockIdx.z;
  const int q0 = blockIdx.x * 128, d0 = blockIdx.y * 128;
  const bf16*  pbm = p + (size_t)b * SQ_ * SK_ + (size_t)q0 * SK_;
  const float* cb  = c + (size_t)b * SK_ * D_ + d0;
  const float* ub  = u + (size_t)b * SQ_ * D_;
  bf16* tb = tt + (size_t)b * SQ_ * D_;

  __shared__ bf16 lA[2][128 * LDT];
  __shared__ bf16 lB[2][128 * LDT];
  const int t = threadIdx.x, lane = t & 31, w = t >> 5;
  const int wm = (w & 1) * 64, wn = (w >> 1) * 32;

  PrefB pa = gload_bf16(pbm, SK_, t);
  PrefF pc = gload_T_f32(cb, D_, t);
  lstore_bf16(lA[0], pa, t);
  lstore_T(lB[0], pc, t);

  v8f acc[4][2] = {};
  constexpr int NS = SK_ / 64;
  for (int it = 0; it < NS; ++it) {
    __syncthreads();
    const int cur = it & 1;
    if (it + 1 < NS) {
      pa = gload_bf16(pbm + (it + 1) * 64, SK_, t);
      pc = gload_T_f32(cb + (size_t)(it + 1) * 64 * D_, D_, t);
    }
    wave_mma(lA[cur], lB[cur], wm, wn, lane, acc);
    if (it + 1 < NS) {
      lstore_bf16(lA[cur ^ 1], pa, t);
      lstore_T(lB[cur ^ 1], pc, t);
    }
  }

  const int l = lane & 15, h = lane >> 4;
#pragma unroll
  for (int mi = 0; mi < 4; ++mi)
#pragma unroll
    for (int ni = 0; ni < 2; ++ni)
#pragma unroll
      for (int r = 0; r < 8; ++r) {
        const int row = q0 + wm + mi * 16 + h * 8 + r;
        const int col = d0 + wn + ni * 16 + l;
        tb[(size_t)row * D_ + col] =
            (bf16)(acc[mi][ni][r] + ub[(size_t)row * D_ + col]);
      }
}

// ---- K4: out = t . W^T + bias ---------------------------------------------

__global__ __launch_bounds__(256) void mqa_k4_linear(
    const bf16* __restrict__ tt, const float* __restrict__ W,
    const float* __restrict__ bias, float* __restrict__ out) {
  const int r0 = blockIdx.x * 128;   // rows over B*SQ = 16384
  const int e0 = blockIdx.y * 128;
  const bf16*  ta = tt + (size_t)r0 * D_;
  const float* Wb = W + (size_t)e0 * D_;

  __shared__ bf16 lA[2][128 * LDT];
  __shared__ bf16 lB[2][128 * LDT];
  const int t = threadIdx.x, lane = t & 31, w = t >> 5;
  const int wm = (w & 1) * 64, wn = (w >> 1) * 32;

  PrefB pa = gload_bf16(ta, D_, t);
  PrefF pb = gload_f32(Wb, D_, t);
  lstore_bf16(lA[0], pa, t);
  lstore_f32(lB[0], pb, t);

  v8f acc[4][2] = {};
  constexpr int NS = D_ / 64;
  for (int it = 0; it < NS; ++it) {
    __syncthreads();
    const int cur = it & 1;
    if (it + 1 < NS) {
      pa = gload_bf16(ta + (it + 1) * 64, D_, t);
      pb = gload_f32(Wb + (it + 1) * 64, D_, t);
    }
    wave_mma(lA[cur], lB[cur], wm, wn, lane, acc);
    if (it + 1 < NS) {
      lstore_bf16(lA[cur ^ 1], pa, t);
      lstore_f32(lB[cur ^ 1], pb, t);
    }
  }

  const int l = lane & 15, h = lane >> 4;
#pragma unroll
  for (int mi = 0; mi < 4; ++mi)
#pragma unroll
    for (int ni = 0; ni < 2; ++ni) {
      const float be = bias[e0 + wn + ni * 16 + l];
#pragma unroll
      for (int r = 0; r < 8; ++r)
        out[(size_t)(r0 + wm + mi * 16 + h * 8 + r) * D_ + (e0 + wn + ni * 16 + l)] =
            acc[mi][ni][r] + be;
    }
}

// ---- launcher --------------------------------------------------------------

extern "C" void kernel_launch(void* const* d_in, const int* in_sizes, int n_in,
                              void* d_out, int out_size, void* d_ws, size_t ws_size,
                              hipStream_t stream) {
  const float* u    = (const float*)d_in[0];
  const float* m    = (const float*)d_in[1];
  const float* c    = (const float*)d_in[2];
  const float* W    = (const float*)d_in[3];
  const float* bias = (const float*)d_in[4];
  float* out = (float*)d_out;

  char* ws = (char*)d_ws;
  float* s = (float*)ws;             ws += (size_t)B_ * SQ_ * SK_ * sizeof(float);
  bf16*  p = (bf16*)ws;              ws += (size_t)B_ * SQ_ * SK_ * sizeof(bf16);
  bf16*  t = (bf16*)ws;              // B*SQ*D bf16

  mqa_k1_scores <<<dim3(SQ_ / 128, SK_ / 128, B_),   256, 0, stream>>>(u, m, s);
  mqa_k2_softmax<<<dim3(SK_ / 32, B_),               256, 0, stream>>>(s, p);
  mqa_k3_ov     <<<dim3(SQ_ / 128, D_ / 128, B_),    256, 0, stream>>>(p, c, u, t);
  mqa_k4_linear <<<dim3((B_ * SQ_) / 128, D_ / 128), 256, 0, stream>>>(t, W, bias, out);
}
